// OccSemanticProjector_49194555408938
// MI455X (gfx1250) — compile-verified, matrix operand order checked by reading, not run on hardware
//
#include <hip/hip_runtime.h>
#include <hip/hip_bf16.h>

typedef __attribute__((ext_vector_type(16))) _Float16 v16h;
typedef __attribute__((ext_vector_type(8)))  float    v8f;

#define HB 224
#define WB 448
#define HW (HB*WB)          // 100352
#define NX 100
#define NY 100
#define NZ 10
#define NVOX (NX*NY*NZ)     // 100000 (divisible by 16)
#define BT 8                // b*t = 1*8
#define NVIEW 6
#define CIN 18
#define COUT 8

// ---------------------------------------------------------------- utilities
__global__ void __launch_bounds__(256) k_zero(float* __restrict__ p, long n) {
  long i = (long)blockIdx.x * 256 + threadIdx.x;
  if (i < n) p[i] = 0.0f;
}

__device__ __forceinline__ void invert4x4(const float m[16], float inv[16]) {
  inv[0]  =  m[5]*m[10]*m[15] - m[5]*m[11]*m[14] - m[9]*m[6]*m[15] + m[9]*m[7]*m[14] + m[13]*m[6]*m[11] - m[13]*m[7]*m[10];
  inv[4]  = -m[4]*m[10]*m[15] + m[4]*m[11]*m[14] + m[8]*m[6]*m[15] - m[8]*m[7]*m[14] - m[12]*m[6]*m[11] + m[12]*m[7]*m[10];
  inv[8]  =  m[4]*m[9]*m[15]  - m[4]*m[11]*m[13] - m[8]*m[5]*m[15] + m[8]*m[7]*m[13] + m[12]*m[5]*m[11] - m[12]*m[7]*m[9];
  inv[12] = -m[4]*m[9]*m[14]  + m[4]*m[10]*m[13] + m[8]*m[5]*m[14] - m[8]*m[6]*m[13] - m[12]*m[5]*m[10] + m[12]*m[6]*m[9];
  inv[1]  = -m[1]*m[10]*m[15] + m[1]*m[11]*m[14] + m[9]*m[2]*m[15] - m[9]*m[3]*m[14] - m[13]*m[2]*m[11] + m[13]*m[3]*m[10];
  inv[5]  =  m[0]*m[10]*m[15] - m[0]*m[11]*m[14] - m[8]*m[2]*m[15] + m[8]*m[3]*m[14] + m[12]*m[2]*m[11] - m[12]*m[3]*m[10];
  inv[9]  = -m[0]*m[9]*m[15]  + m[0]*m[11]*m[13] + m[8]*m[1]*m[15] - m[8]*m[3]*m[13] - m[12]*m[1]*m[11] + m[12]*m[3]*m[9];
  inv[13] =  m[0]*m[9]*m[14]  - m[0]*m[10]*m[13] - m[8]*m[1]*m[14] + m[8]*m[2]*m[13] + m[12]*m[1]*m[10] - m[12]*m[2]*m[9];
  inv[2]  =  m[1]*m[6]*m[15]  - m[1]*m[7]*m[14]  - m[5]*m[2]*m[15] + m[5]*m[3]*m[14] + m[13]*m[2]*m[7]  - m[13]*m[3]*m[6];
  inv[6]  = -m[0]*m[6]*m[15]  + m[0]*m[7]*m[14]  + m[4]*m[2]*m[15] - m[4]*m[3]*m[14] - m[12]*m[2]*m[7]  + m[12]*m[3]*m[6];
  inv[10] =  m[0]*m[5]*m[15]  - m[0]*m[7]*m[13]  - m[4]*m[1]*m[15] + m[4]*m[3]*m[13] + m[12]*m[1]*m[7]  - m[12]*m[3]*m[5];
  inv[14] = -m[0]*m[5]*m[14]  + m[0]*m[6]*m[13]  + m[4]*m[1]*m[14] - m[4]*m[2]*m[13] - m[12]*m[1]*m[6]  + m[12]*m[2]*m[5];
  inv[3]  = -m[1]*m[6]*m[11]  + m[1]*m[7]*m[10]  + m[5]*m[2]*m[11] - m[5]*m[3]*m[10] - m[9]*m[2]*m[7]   + m[9]*m[3]*m[6];
  inv[7]  =  m[0]*m[6]*m[11]  - m[0]*m[7]*m[10]  - m[4]*m[2]*m[11] + m[4]*m[3]*m[10] + m[8]*m[2]*m[7]   - m[8]*m[3]*m[6];
  inv[11] = -m[0]*m[5]*m[11]  + m[0]*m[7]*m[9]   + m[4]*m[1]*m[11] - m[4]*m[3]*m[9]  - m[8]*m[1]*m[7]   + m[8]*m[3]*m[5];
  inv[15] =  m[0]*m[5]*m[10]  - m[0]*m[6]*m[9]   - m[4]*m[1]*m[10] + m[4]*m[2]*m[9]  + m[8]*m[1]*m[6]   - m[8]*m[2]*m[5];
  float det = m[0]*inv[0] + m[1]*inv[4] + m[2]*inv[8] + m[3]*inv[12];
  float d = 1.0f / det;
  #pragma unroll
  for (int i = 0; i < 16; ++i) inv[i] *= d;
}

// One thread per (t, view): invert 4x4 c2w, keep rows 0..2 (the only rows used).
__global__ void k_w2c(const float* __restrict__ c2w, float* __restrict__ w2c_ws) {
  int i = threadIdx.x;
  if (i >= BT * NVIEW) return;
  float m[16], inv[16];
  #pragma unroll
  for (int j = 0; j < 16; ++j) m[j] = c2w[i * 16 + j];
  invert4x4(m, inv);
  #pragma unroll
  for (int j = 0; j < 12; ++j) w2c_ws[i * 12 + j] = inv[j];
}

// ------------------------------------------------- WMMA projection + softmax + alpha
// One wave handles a tile of 16 voxel instances. D(16x16) = Wpad(16x32) x SemPad(32x16) + Bias.
// A layout (16-bit, 16x32): lane L<16 -> row M=L, halves[0..7]=K0..7, halves[8..15]=K16..23;
//                           lane L>=16 -> row M=L-16, halves[0..7]=K8..15, halves[8..15]=K24..31.
// B mirrors A with N (voxel column) in place of M. D: lane L<16 holds column N=L, VGPR r -> row M=r.
__global__ void __launch_bounds__(256)
k_proj_softmax_alpha(const float* __restrict__ occ, const float* __restrict__ sem,
                     const float* __restrict__ pw,  const float* __restrict__ pb,
                     float* __restrict__ sem_ws, float* __restrict__ alpha_ws)
{
  const int  tid  = blockIdx.x * 256 + threadIdx.x;
  const int  wave = tid >> 5;
  const int  lane = threadIdx.x & 31;
  const int  col  = lane & 15;       // M for A, N (voxel) for B/D
  const int  hi   = lane >> 4;       // K-half select
  const long g    = (long)wave * 16 + col;   // global voxel instance (BT*NVOX total, /16 exact)
  const int  bt   = (int)(g / NVOX);
  const int  v    = (int)(g % NVOX);

  v16h A, B;
  #pragma unroll
  for (int j = 0; j < 8; ++j) {
    int k = hi * 8 + j;
    A[j] = (_Float16)((col < COUT && k < CIN) ? pw[col * CIN + k] : 0.0f);
  }
  #pragma unroll
  for (int j = 0; j < 8; ++j) {
    int k = 16 + hi * 8 + j;
    A[8 + j] = (_Float16)((col < COUT && k < CIN) ? pw[col * CIN + k] : 0.0f);
  }
  const float* semb = sem + (long)bt * CIN * NVOX + v;
  #pragma unroll
  for (int j = 0; j < 8; ++j) {
    int k = hi * 8 + j;
    B[j] = (_Float16)((k < CIN) ? semb[(long)k * NVOX] : 0.0f);
  }
  #pragma unroll
  for (int j = 0; j < 8; ++j) {
    int k = 16 + hi * 8 + j;
    B[8 + j] = (_Float16)((k < CIN) ? semb[(long)k * NVOX] : 0.0f);
  }
  v8f C;
  #pragma unroll
  for (int r = 0; r < 8; ++r) C[r] = (hi == 0) ? pb[r] : 0.0f;

  // (neg_a, A, neg_b, B, c_mod, C, reuse_a, reuse_b)
  v8f D = __builtin_amdgcn_wmma_f32_16x16x32_f16(false, A, false, B, (short)0, C, false, false);

  if (hi == 0) {  // lanes 0..15 each own the full 8-channel column for voxel v
    float m = D[0];
    #pragma unroll
    for (int r = 1; r < 8; ++r) m = fmaxf(m, D[r]);
    float e[8], s = 0.0f;
    #pragma unroll
    for (int r = 0; r < 8; ++r) { e[r] = __expf(D[r] - m); s += e[r]; }
    float inv = 1.0f / s;
    float* dst = sem_ws + ((long)bt * NVOX + v) * 8;
    #pragma unroll
    for (int r = 0; r < 8; ++r) dst[r] = e[r] * inv;
    // alpha = 1 - exp(-softplus(x)) == sigmoid(x)
    float x = occ[(long)bt * NVOX + v];
    alpha_ws[(long)bt * NVOX + v] = 1.0f / (1.0f + __expf(-x));
  }
}

// ------------------------------------------------- bilinear splat (atomics into out/den)
__global__ void __launch_bounds__(256)
k_scatter(const float* __restrict__ alpha_ws, const float* __restrict__ sem_ws,
          const float* __restrict__ w2c_ws,   const float* __restrict__ intr,
          const float* __restrict__ ego,      float* __restrict__ out,
          float* __restrict__ den)
{
  long g = (long)blockIdx.x * 256 + threadIdx.x;
  if (g >= (long)BT * NVOX) return;
  int bt = (int)(g / NVOX), n = (int)(g % NVOX);
  int iz = n / (NX * NY);
  int rem = n % (NX * NY);
  int iy = rem / NX, ix = rem % NX;
  float px = -39.6f + 0.8f * ix;
  float py = -39.6f + 0.8f * iy;
  float pz = -1.6f  + 0.8f * iz;
  // world = ego @ [px,py,pz,1] (full homogeneous, ego bottom row is perturbed)
  float w0 = ego[0]  * px + ego[1]  * py + ego[2]  * pz + ego[3];
  float w1 = ego[4]  * px + ego[5]  * py + ego[6]  * pz + ego[7];
  float w2 = ego[8]  * px + ego[9]  * py + ego[10] * pz + ego[11];
  float w3 = ego[12] * px + ego[13] * py + ego[14] * pz + ego[15];

  float alpha = alpha_ws[g];
  float s[8];
  const float* sp = sem_ws + g * 8;
  #pragma unroll
  for (int c = 0; c < 8; ++c) s[c] = sp[c];

  for (int view = 0; view < NVIEW; ++view) {
    const float* M = w2c_ws + (bt * NVIEW + view) * 12;
    float cx_ = M[0] * w0 + M[1] * w1 + M[2]  * w2 + M[3]  * w3;
    float cy_ = M[4] * w0 + M[5] * w1 + M[6]  * w2 + M[7]  * w3;
    float cz_ = M[8] * w0 + M[9] * w1 + M[10] * w2 + M[11] * w3;
    const float* K = intr + view * 4;
    float zok = (cz_ > 0.001f) ? cz_ : 1.0f;
    float u  = cx_ * K[0] / zok + K[2];
    float vv = cy_ * K[1] / zok + K[3];
    bool base = (cz_ > 0.001f) & (alpha > 0.0001f) &
                (u >= -1.0f) & (u <= (float)WB) & (vv >= -1.0f) & (vv <= (float)HB);
    if (!base) continue;
    int x0 = (int)floorf(u), y0 = (int)floorf(vv);
    float* denp = den + (long)(bt * NVIEW + view) * HW;
    float* outp = out + (long)(bt * NVIEW + view) * COUT * HW;
    #pragma unroll
    for (int tap = 0; tap < 4; ++tap) {
      int xn = x0 + (tap & 1), yn = y0 + (tap >> 1);
      if (xn < 0 || xn >= WB || yn < 0 || yn >= HB) continue;
      float du = fminf(fmaxf(1.0f - fabsf(u  - (float)xn), 0.0f), 1.0f);
      float dv = fminf(fmaxf(1.0f - fabsf(vv - (float)yn), 0.0f), 1.0f);
      float wgt = du * dv * alpha;
      int idx = yn * WB + xn;
      atomicAdd(denp + idx, wgt);
      #pragma unroll
      for (int c = 0; c < 8; ++c) atomicAdd(outp + c * HW + idx, s[c] * wgt);
    }
  }
}

// ------------------------------------------------- final normalize: out /= max(den,1e-6)
__global__ void __launch_bounds__(256)
k_norm(float* __restrict__ out, const float* __restrict__ den) {
  long i = (long)blockIdx.x * 256 + threadIdx.x;
  const long total = (long)BT * NVIEW * COUT * HW;
  if (i >= total) return;
  long pix = i % HW;
  long btv = i / ((long)COUT * HW);
  out[i] = out[i] / fmaxf(den[btv * HW + pix], 1e-6f);
}

// ---------------------------------------------------------------- launch
extern "C" void kernel_launch(void* const* d_in, const int* in_sizes, int n_in,
                              void* d_out, int out_size, void* d_ws, size_t ws_size,
                              hipStream_t stream) {
  const float* occ  = (const float*)d_in[0];  // (1,8,1,10,100,100)
  const float* sem  = (const float*)d_in[1];  // (1,8,18,10,100,100)
  const float* intr = (const float*)d_in[2];  // (1,6,4)
  const float* c2w  = (const float*)d_in[3];  // (1,8,6,4,4)
  const float* ego  = (const float*)d_in[4];  // (1,4,4)
  const float* pw   = (const float*)d_in[5];  // (8,18)
  const float* pb   = (const float*)d_in[6];  // (8,)
  float* out = (float*)d_out;                 // (1,8,6,8,224,448)

  // Workspace layout (floats): den | w2c | sem_ws | alpha_ws
  float* den      = (float*)d_ws;                                  // BT*NVIEW*HW
  float* w2c_ws   = den      + (long)BT * NVIEW * HW;              // BT*NVIEW*12
  float* sem_ws   = w2c_ws   + (long)BT * NVIEW * 12;              // BT*NVOX*8
  float* alpha_ws = sem_ws   + (long)BT * NVOX * COUT;             // BT*NVOX

  const long n_out = (long)BT * NVIEW * COUT * HW;   // 38,535,168
  const long n_den = (long)BT * NVIEW * HW;          //  4,816,896

  k_zero<<<(int)((n_out + 255) / 256), 256, 0, stream>>>(out, n_out);
  k_zero<<<(int)((n_den + 255) / 256), 256, 0, stream>>>(den, n_den);
  k_w2c<<<1, 64, 0, stream>>>(c2w, w2c_ws);

  // 50,000 waves (one per 16-voxel tile), 8 waves per 256-thread block
  k_proj_softmax_alpha<<<(BT * NVOX / 16) / 8, 256, 0, stream>>>(
      occ, sem, pw, pb, sem_ws, alpha_ws);

  k_scatter<<<(int)(((long)BT * NVOX + 255) / 256), 256, 0, stream>>>(
      alpha_ws, sem_ws, w2c_ws, intr, ego, out, den);

  k_norm<<<(int)((n_out + 255) / 256), 256, 0, stream>>>(out, den);
}
